// Euler_25726854103314
// MI455X (gfx1250) — matrix operand order repaired and checked by reading, not compiled, not run on hardware
//
#include <hip/hip_runtime.h>
#include <stdint.h>

typedef __attribute__((ext_vector_type(16))) _Float16 v16h;
typedef __attribute__((ext_vector_type(8)))  _Float16 v8h;
typedef __attribute__((ext_vector_type(8)))  float    v8f;

constexpr int   Bd = 1024, Ld = 512, Sd = 64, Ud = 32, Hd = 512, SU = 96;
constexpr int   BM = 16;          // batch rows per workgroup
constexpr float DT = 0.1f;

// LDS layout:
//   w1t : [H][SU]  f16  (W1 transposed)      =  98304 B
//   w2t : [S][H]   f16  (W2 transposed)      =  65536 B
//   zbuf: [BM][SU] f16  (concat [state,u])   =   3072 B
//   hbuf: [BM][H]  f16  (tanh hidden)        =  16384 B
//   stat: [BM][S]  f32                       =   4096 B
//   b1s : [H] f32 (2048 B), b2s : [S] f32 (256 B)
//   ustg: [2][BM][U] f32 async staging       =   4096 B
constexpr size_t LDS_BYTES =
    (size_t)(Hd * SU + Sd * Hd + BM * SU + BM * Hd) * 2 +
    (size_t)(BM * Sd + Hd + Sd + 2 * BM * Ud) * 4;   // = 193,792 B (CDNA5 320KB LDS)

union Frag {
    v16h v;
    struct { v8h lo, hi; } p;
};

// A-matrix fragment (16-bit, 16x32): per-lane 16 contiguous halves at
// +0 and +8 halves (caller pre-applies the 0/16 half offset for lanes>=16).
__device__ __forceinline__ v16h load_a(const _Float16* p) {
    Frag f;
    f.p.lo = *(const v8h*)(p);
    f.p.hi = *(const v8h*)(p + 8);
    return f.v;
}

__device__ __forceinline__ v8f wmma(v16h a, v16h b, v8f c) {
    return __builtin_amdgcn_wmma_f32_16x16x32_f16(
        /*neg_a=*/false, a, /*neg_b=*/false, b,
        /*c_mod=*/(short)0, c, /*reuse_a=*/false, /*reuse_b=*/false);
}

// Branch-free tanh: prefer CDNA5 hardware v_tanh_f32; else exp2/rcp formula.
// tanh(|x|) = 1 - 2/(exp2(2*log2e*|x|) + 1), sign restored via copysign.
__device__ __forceinline__ float fast_tanh(float x) {
#if __has_builtin(__builtin_amdgcn_tanhf)
    return __builtin_amdgcn_tanhf(x);
#else
    float ax = __builtin_fabsf(x);
    float e  = __builtin_amdgcn_exp2f(2.8853900817779268f * ax); // 2*log2(e)
    float r  = __builtin_amdgcn_rcpf(e + 1.0f);
    float t  = __builtin_fmaf(-2.0f, r, 1.0f);
    return __builtin_copysignf(t, x);
#endif
}

// Async global -> LDS copy, 16B per lane (ASYNCcnt-tracked, CDNA5).
// vdst VGPR carries the LDS byte address = low 32 bits of the generic pointer.
__device__ __forceinline__ void async_u_load(const float* gp, float* lp) {
    asm volatile("global_load_async_to_lds_b128 %0, %1, off"
                 :: "v"((uint32_t)(uintptr_t)lp), "v"(gp)
                 : "memory");
}
__device__ __forceinline__ void wait_asynccnt0() {
    asm volatile("s_wait_asynccnt 0x0" ::: "memory");
}

__global__ __launch_bounds__(128, 1)
void euler_mlp_kernel(const float* __restrict__ init,
                      const float* __restrict__ ctrl,
                      const float* __restrict__ W1,
                      const float* __restrict__ b1,
                      const float* __restrict__ W2,
                      const float* __restrict__ b2,
                      float* __restrict__ out) {
    extern __shared__ char smem[];
    _Float16* w1t  = (_Float16*)smem;            // [H][SU]
    _Float16* w2t  = w1t  + Hd * SU;             // [S][H]
    _Float16* zbuf = w2t  + Sd * Hd;             // [BM][SU]
    _Float16* hbuf = zbuf + BM * SU;             // [BM][H]
    float*    stat = (float*)(hbuf + BM * Hd);   // [BM][S]
    float*    b1s  = stat + BM * Sd;             // [H]
    float*    b2s  = b1s + Hd;                   // [S]
    float*    ustg = b2s + Sd;                   // [2][BM][U]

    const int tid  = threadIdx.x;
    const int lane = tid & 31;
    const int wv   = tid >> 5;                   // wave 0..3
    const int col  = lane & 15;                  // N / column within 16-tile
    const int hb   = (lane & 16) ? 8 : 0;        // C/D row offset (lanes 16-31)
    const int khbA = (lane & 16) ? 16 : 0;       // A-frag K start (0/16)
    const int kstB = (lane & 16) ? 16 : 0;       // B-frag K start (0/16)
    const int row0 = blockIdx.x * BM;

    // Per-thread u_t slice: 4 floats = 16 B (one async b128 per lane).
    const int ur  = tid >> 3;                    // 0..15
    const int uu4 = (tid & 7) * 4;               // 0..28
    const float* ug0 = ctrl + ((size_t)(row0 + ur) * Ld) * Ud + uu4;

    // Kick off the async copy of u_0 while we stage weights.
    async_u_load(ug0, ustg + ur * Ud + uu4);

    // ---- one-time preload: weights (transposed, f16), biases, state ----
    for (int i = tid; i < Hd * SU; i += blockDim.x) {
        int h = i / SU, k = i - h * SU;
        w1t[i] = (_Float16)W1[k * Hd + h];
    }
    for (int i = tid; i < Sd * Hd; i += blockDim.x) {
        int s = i / Hd, k = i - s * Hd;
        w2t[i] = (_Float16)W2[k * Sd + s];
    }
    for (int i = tid; i < Hd; i += blockDim.x) b1s[i] = b1[i];
    for (int i = tid; i < Sd; i += blockDim.x) b2s[i] = b2[i];
    for (int i = tid; i < BM * Sd; i += blockDim.x) stat[i] = init[row0 * Sd + i];
    __syncthreads();

    const v8f zero8 = (v8f){0.f, 0.f, 0.f, 0.f, 0.f, 0.f, 0.f, 0.f};

    for (int t = 0; t < Ld; ++t) {
        // ---- build z = [state | u_t] in f16 ----
        wait_asynccnt0();                        // u_t staged in LDS
        for (int i = tid; i < BM * Sd; i += blockDim.x) {
            int r = i >> 6, s = i & 63;
            zbuf[r * SU + s] = (_Float16)stat[i];
        }
        {
            const float* us = ustg + (t & 1) * (BM * Ud) + ur * Ud + uu4;
            float4 uv = *(const float4*)us;      // each lane reads what it loaded
            _Float16* zp = zbuf + ur * SU + Sd + uu4;
            zp[0] = (_Float16)uv.x; zp[1] = (_Float16)uv.y;
            zp[2] = (_Float16)uv.z; zp[3] = (_Float16)uv.w;
            if (t + 1 < Ld)                      // pipeline u_{t+1}
                async_u_load(ug0 + (size_t)(t + 1) * Ud,
                             ustg + ((t + 1) & 1) * (BM * Ud) + ur * Ud + uu4);
        }
        __syncthreads();

        // ---- GEMM1: h = tanh(z @ W1 + b1) ----
        // Zero-C start + 8 independent accumulator chains; bias folded into
        // the tanh epilogue so no per-step bias splat into accumulators.
        {
            const _Float16* za = zbuf + (lane & 15) * SU + khbA;
            v16h a0 = load_a(za);
            v16h a1 = load_a(za + 32);
            v16h a2 = load_a(za + 64);

            const _Float16* wb0 = w1t + ((wv * 8) * 16 + col) * SU + kstB;
            v8f acc[8];
            #pragma unroll
            for (int i = 0; i < 8; ++i)
                acc[i] = wmma(a0, *(const v16h*)(wb0 + i * 16 * SU), zero8);
            #pragma unroll
            for (int i = 0; i < 8; ++i)
                acc[i] = wmma(a1, *(const v16h*)(wb0 + i * 16 * SU + 32), acc[i]);
            #pragma unroll
            for (int i = 0; i < 8; ++i)
                acc[i] = wmma(a2, *(const v16h*)(wb0 + i * 16 * SU + 64), acc[i]);

            #pragma unroll
            for (int i = 0; i < 8; ++i) {
                const int nt = wv * 8 + i;
                const float bias = b1s[nt * 16 + col];
                #pragma unroll
                for (int r = 0; r < 8; ++r)
                    hbuf[(r + hb) * Hd + nt * 16 + col] =
                        (_Float16)fast_tanh(acc[i][r] + bias);
            }
        }
        __syncthreads();

        // ---- GEMM2: diff = h @ W2 + b2; state += DT*diff; emit state ----
        {
            const _Float16* ha = hbuf + (lane & 15) * Hd + khbA;
            const _Float16* wb = w2t + (wv * 16 + col) * Hd + kstB;
            // 4 round-robin partial accumulators -> 4 independent WMMA chains.
            v8f c0 = wmma(load_a(ha +  0), *(const v16h*)(wb +  0), zero8);
            v8f c1 = wmma(load_a(ha + 32), *(const v16h*)(wb + 32), zero8);
            v8f c2 = wmma(load_a(ha + 64), *(const v16h*)(wb + 64), zero8);
            v8f c3 = wmma(load_a(ha + 96), *(const v16h*)(wb + 96), zero8);
            #pragma unroll
            for (int kt = 4; kt < 16; kt += 4) {
                c0 = wmma(load_a(ha + (kt + 0) * 32), *(const v16h*)(wb + (kt + 0) * 32), c0);
                c1 = wmma(load_a(ha + (kt + 1) * 32), *(const v16h*)(wb + (kt + 1) * 32), c1);
                c2 = wmma(load_a(ha + (kt + 2) * 32), *(const v16h*)(wb + (kt + 2) * 32), c2);
                c3 = wmma(load_a(ha + (kt + 3) * 32), *(const v16h*)(wb + (kt + 3) * 32), c3);
            }
            v8f c = (c0 + c1) + (c2 + c3);
            const float bias2 = b2s[wv * 16 + col];
            #pragma unroll
            for (int r = 0; r < 8; ++r) {
                const int row = r + hb;
                const int sc  = wv * 16 + col;
                float ns = __builtin_fmaf(DT, c[r] + bias2, stat[row * Sd + sc]);
                stat[row * Sd + sc] = ns;
                out[((size_t)(row0 + row) * Ld + t) * Sd + sc] = ns;
            }
        }
        __syncthreads();
    }
}

extern "C" void kernel_launch(void* const* d_in, const int* in_sizes, int n_in,
                              void* d_out, int out_size, void* d_ws, size_t ws_size,
                              hipStream_t stream) {
    (void)in_sizes; (void)n_in; (void)d_ws; (void)ws_size; (void)out_size;
    const float* init = (const float*)d_in[0];
    const float* ctrl = (const float*)d_in[1];
    const float* W1   = (const float*)d_in[2];
    const float* b1   = (const float*)d_in[3];
    const float* W2   = (const float*)d_in[4];
    const float* b2   = (const float*)d_in[5];
    float* out = (float*)d_out;

    euler_mlp_kernel<<<dim3(Bd / BM), dim3(128), LDS_BYTES, stream>>>(
        init, ctrl, W1, b1, W2, b2, out);
}
